// ParallelTransformerBlock_50070728736955
// MI455X (gfx1250) — matrix operand order repaired and checked by reading, not compiled, not run on hardware
//
#include <hip/hip_runtime.h>
#include <hip/hip_bf16.h>
#include <math.h>

// ---------------------------------------------------------------------------
// Types
// ---------------------------------------------------------------------------
typedef unsigned short u16;
typedef __attribute__((ext_vector_type(16))) __bf16 bf16x16;  // WMMA A/B frag
typedef __attribute__((ext_vector_type(8)))  float  floatx8;  // WMMA C/D frag
typedef __attribute__((ext_vector_type(4)))  unsigned int uint4v;
typedef __attribute__((ext_vector_type(4)))  int int4v;

// gfx1250 async global->LDS path (ASYNCcnt), guarded so the build never breaks.
// Toolchain signature (from round-2 diagnostic): (V4i as1*, V4i as3*, Ii, Ii)
#if defined(__has_builtin)
#if __has_builtin(__builtin_amdgcn_global_load_async_to_lds_b128) && \
    __has_builtin(__builtin_amdgcn_s_wait_asynccnt)
#define HAVE_ASYNC_LDS 1
#endif
#endif
#ifndef HAVE_ASYNC_LDS
#define HAVE_ASYNC_LDS 0
#endif

#if HAVE_ASYNC_LDS
typedef __attribute__((address_space(1))) int4v gint4v;  // global int4
typedef __attribute__((address_space(3))) int4v lint4v;  // LDS int4
#define ASYNC_B128(gp, lp)                                                  \
  __builtin_amdgcn_global_load_async_to_lds_b128((gint4v*)(gp),             \
                                                 (lint4v*)(lp), 0, 0)
#define ASYNC_WAIT() __builtin_amdgcn_s_wait_asynccnt(0)
#else
#define ASYNC_B128(gp, lp) (*(uint4v*)(lp) = *(const uint4v*)(gp))
#define ASYNC_WAIT() ((void)0)
#endif

__device__ __forceinline__ u16 f2bf(float f) {
  union { float f; unsigned u; } x; x.f = f;
  unsigned r = x.u + 0x7FFFu + ((x.u >> 16) & 1u);   // round-to-nearest-even
  return (u16)(r >> 16);
}
__device__ __forceinline__ float bf2f(u16 h) {
  union { unsigned u; float f; } x; x.u = ((unsigned)h) << 16;
  return x.f;
}

union FragU { bf16x16 v; uint4v q[2]; };

// A-matrix fragment 16(M) x 32(K) bf16, source row-major [m][k], ld in u16.
// ISA layout: lanes 0-15 M=0..15 / lanes 16-31 same M; VGPR0-3 K = kh*8+e,
// VGPR4-7 K = 16+kh*8+(e-8), kh = lane>>4.
__device__ __forceinline__ bf16x16 load_a_frag(const u16* base, int ld, int lane) {
  const u16* p = base + (lane & 15) * ld + ((lane >> 4) << 3);
  FragU f;
  f.q[0] = *(const uint4v*)(p);
  f.q[1] = *(const uint4v*)(p + 16);
  return f.v;
}

// B-matrix fragment 32(K) x 16(N) bf16, source stored transposed [n][k], ld in
// u16. ISA layout: VGPR0-3 K = kh*8+e, VGPR4-7 K = 16+kh*8+(e-8), kh = lane>>4.
__device__ __forceinline__ bf16x16 load_b_frag(const u16* base, int ld, int lane) {
  const u16* p = base + (lane & 15) * ld + ((lane >> 4) << 3);
  FragU f;
  f.q[0] = *(const uint4v*)(p);
  f.q[1] = *(const uint4v*)(p + 16);
  return f.v;
}

__device__ __forceinline__ floatx8 wmma_bf16(bf16x16 a, bf16x16 b, floatx8 c) {
  return __builtin_amdgcn_wmma_f32_16x16x32_bf16(false, a, false, b, (short)0, c,
                                                 false, false);
}

// ---------------------------------------------------------------------------
// Elementwise kernels
// ---------------------------------------------------------------------------
__global__ void cvt_f32_bf16(const float* __restrict__ in, u16* __restrict__ out,
                             long n) {
  long i = (long)blockIdx.x * blockDim.x + threadIdx.x;
  long stride = (long)gridDim.x * blockDim.x;
  for (; i < n; i += stride) out[i] = f2bf(in[i]);
}

// silu(gate) * x from fused ff projection (bf16 in, bf16 out)
__global__ void silu_mul(const u16* __restrict__ pff, u16* __restrict__ act) {
  const long n = 4096l * 8192l;
  long idx = (long)blockIdx.x * blockDim.x + threadIdx.x;
  long stride = (long)gridDim.x * blockDim.x;
  for (; idx < n; idx += stride) {
    long i = idx >> 13, j = idx & 8191;
    float xx = bf2f(pff[i * 16384 + j]);
    float g  = bf2f(pff[i * 16384 + 8192 + j]);
    act[idx] = f2bf(xx * (g / (1.f + __expf(-g))));
  }
}

// xpos-scaled rotary: q (scaled by d^-1/2, *xscale) and k (*1/xscale), v copy.
// grid (2048 positions, 2 batches), 128 threads = head dim.
__global__ void rotary_kernel(const float* __restrict__ pqkv,
                              u16* __restrict__ q_rot, u16* __restrict__ k_rot,
                              u16* __restrict__ v_b) {
  const int ni = blockIdx.x, bi = blockIdx.y, j = threadIdx.x;  // j in [0,128)
  const int half = j & 63;
  const float invf = powf(10000.f, -(float)half / 64.f);
  const float pos = (float)ni * invf;
  const float c = cosf(pos), s = sinf(pos);
  const float bs = (2.f * half + 51.2f) / 179.2f;     // (2i + .4d)/(1.4d)
  const float pw = ((float)ni - 1024.f) / 512.f;
  const float xs = powf(bs, pw);
  const float* row = pqkv + (long)(bi * 2048 + ni) * 2304;
  // k (cols 2048..2175) with inverse scale
  {
    float kv = row[2048 + j];
    float kr = (j < 64) ? -row[2048 + j + 64] : row[2048 + j - 64];
    float ixs = 1.f / xs;
    k_rot[(long)(bi * 2048 + ni) * 128 + j] = f2bf((kv * c + kr * s) * ixs);
  }
  // v (cols 2176..2303)
  v_b[(long)(bi * 2048 + ni) * 128 + j] = f2bf(row[2176 + j]);
  // q: 16 heads, pre-scaled by 128^-0.5
  const float qs = 0.08838834764831845f;
#pragma unroll 4
  for (int h = 0; h < 16; ++h) {
    float qv = row[h * 128 + j] * qs;
    float qr = ((j < 64) ? -row[h * 128 + j + 64] : row[h * 128 + j - 64]) * qs;
    q_rot[((long)((bi * 16 + h) * 2048 + ni)) * 128 + j] =
        f2bf((qv * c + qr * s) * xs);
  }
}

// ---------------------------------------------------------------------------
// bf16 WMMA GEMM: C[M,N] = A[M,K] @ B[K,N], A row-major (lda=K), B row-major
// (ldb). Block tile 128x128x64 (two WMMA K-steps per stage), 8 waves each
// computing 32x64. A tile staged via async global->LDS; B transposed on stage.
// EPI: 0 = f32 store, 1 = bf16 store, 2 = f32 accumulate (+=)
// ---------------------------------------------------------------------------
template <int EPI>
__global__ __launch_bounds__(256) void gemm_bf16(
    const u16* __restrict__ A, const u16* __restrict__ B, void* __restrict__ Cv,
    int K, int ldb, int ldc) {
  __shared__ __align__(16) u16 As[128][64];  // 16KB, row-major [m][k]
  __shared__ __align__(16) u16 Bs[128][64];  // 16KB, transposed [n][k]
  const int tid = threadIdx.x;
  const int lane = tid & 31;
  const int w = tid >> 5;
  const int wr = w & 3;   // 32-row slab
  const int wc = w >> 2;  // 64-col slab
  const int row0 = blockIdx.y * 128;
  const int col0 = blockIdx.x * 128;

  floatx8 z = {0.f, 0.f, 0.f, 0.f, 0.f, 0.f, 0.f, 0.f};
  floatx8 acc[2][4];
#pragma unroll
  for (int m = 0; m < 2; ++m)
#pragma unroll
    for (int n = 0; n < 4; ++n) acc[m][n] = z;

  for (int kt = 0; kt < K; kt += 64) {
    // stage A tile (128x64): 1024 16B chunks, async direct-to-LDS
#pragma unroll
    for (int i = 0; i < 4; ++i) {
      int c = tid + (i << 8);
      int r = c >> 3, kc = (c & 7) << 3;
      ASYNC_B128(A + (long)(row0 + r) * K + kt + kc, &As[r][kc]);
    }
    // stage B tile (64x128) transposed into Bs[n][k]
#pragma unroll
    for (int i = 0; i < 4; ++i) {
      int c = tid + (i << 8);
      int kk = c >> 4, nc = (c & 15) << 3;
      uint4v t = *(const uint4v*)(B + (long)(kt + kk) * ldb + col0 + nc);
      const u16* u = (const u16*)&t;
#pragma unroll
      for (int j = 0; j < 8; ++j) Bs[nc + j][kk] = u[j];
    }
    if (kt + 64 < K) {  // gfx1250 global_prefetch_b8 on next tiles
      __builtin_prefetch(A + (long)(row0 + (tid >> 3)) * K + kt + 64, 0, 0);
      __builtin_prefetch(B + (long)(kt + 64 + (tid >> 4)) * ldb + col0, 0, 0);
    }
    ASYNC_WAIT();
    __syncthreads();

#pragma unroll
    for (int ks = 0; ks < 2; ++ks) {
      bf16x16 af[2], bfr[4];
#pragma unroll
      for (int m = 0; m < 2; ++m)
        af[m] = load_a_frag(&As[wr * 32 + m * 16][ks * 32], 64, lane);
#pragma unroll
      for (int n = 0; n < 4; ++n)
        bfr[n] = load_b_frag(&Bs[wc * 64 + n * 16][ks * 32], 64, lane);
#pragma unroll
      for (int m = 0; m < 2; ++m)
#pragma unroll
        for (int n = 0; n < 4; ++n)
          acc[m][n] = wmma_bf16(af[m], bfr[n], acc[m][n]);
    }
    __syncthreads();
  }

  // epilogue (C layout: VGPR r, lanes0-15 -> M=r, lanes16-31 -> M=r+8; N=lane%16)
#pragma unroll
  for (int m = 0; m < 2; ++m)
#pragma unroll
    for (int n = 0; n < 4; ++n)
#pragma unroll
      for (int r = 0; r < 8; ++r) {
        int row = row0 + wr * 32 + m * 16 + r + ((lane >> 4) << 3);
        int col = col0 + wc * 64 + n * 16 + (lane & 15);
        long off = (long)row * ldc + col;
        if (EPI == 0)      ((float*)Cv)[off] = acc[m][n][r];
        else if (EPI == 1) ((u16*)Cv)[off] = f2bf(acc[m][n][r]);
        else               ((float*)Cv)[off] += acc[m][n][r];
      }
}

// ---------------------------------------------------------------------------
// Causal multi-query flash attention, bf16 WMMA, f32 softmax state.
// grid (qblock=16, head=16, batch=2), 256 threads; wave w owns 16 q rows.
// q_rot: [b*h][n][128], k_rot/v: [b][n][128], o: [b*n][h*128] bf16
// ---------------------------------------------------------------------------
__global__ __launch_bounds__(256) void attn_kernel(
    const u16* __restrict__ q, const u16* __restrict__ k,
    const u16* __restrict__ v, u16* __restrict__ o) {
  __shared__ __align__(16) u16 k_lds[64][128];   // [j][d] == Bt for q.kT
  __shared__ __align__(16) u16 vT_lds[128][64];  // [d][j] == Bt for P.V
  __shared__ __align__(16) u16 p_lds[8][16][64]; // per-wave P staging
  const int tid = threadIdx.x, lane = tid & 31, w = tid >> 5;
  const int qb = blockIdx.x, h = blockIdx.y, b = blockIdx.z;

  const u16* qptr = q + ((long)((b * 16 + h) * 2048 + qb * 128 + w * 16)) * 128;
  bf16x16 qf[4];
#pragma unroll
  for (int ks = 0; ks < 4; ++ks) qf[ks] = load_a_frag(qptr + ks * 32, 128, lane);

  const u16* kbase = k + (long)b * 2048 * 128;
  const u16* vbase = v + (long)b * 2048 * 128;

  floatx8 z = {0.f, 0.f, 0.f, 0.f, 0.f, 0.f, 0.f, 0.f};
  floatx8 oacc[8];
  float mrun[8], lrun[8];
#pragma unroll
  for (int i = 0; i < 8; ++i) { oacc[i] = z; mrun[i] = -3.0e38f; lrun[i] = 0.f; }

  const int nkb = 2 * (qb + 1);  // 64-wide key blocks, causal bound
  for (int kb = 0; kb < nkb; ++kb) {
    // K tile 64x128 is consumed in memory layout -> async direct-to-LDS
#pragma unroll
    for (int i = 0; i < 4; ++i) {
      int c = tid + (i << 8);
      int r = c >> 4, dc = (c & 15) << 3;
      ASYNC_B128(kbase + (long)(kb * 64 + r) * 128 + dc, &k_lds[r][dc]);
    }
    // V tile transposed on stage (needs [d][j] for the P@V B-operand)
#pragma unroll
    for (int i = 0; i < 4; ++i) {
      int c = tid + (i << 8);
      int r = c >> 4, dc = (c & 15) << 3;
      uint4v t = *(const uint4v*)(vbase + (long)(kb * 64 + r) * 128 + dc);
      const u16* u = (const u16*)&t;
#pragma unroll
      for (int j = 0; j < 8; ++j) vT_lds[dc + j][r] = u[j];
    }
    ASYNC_WAIT();
    __syncthreads();

    // S = q @ kT : 16 x 64 in four 16x16 accums
    floatx8 s[4] = {z, z, z, z};
#pragma unroll
    for (int nt = 0; nt < 4; ++nt)
#pragma unroll
      for (int ks = 0; ks < 4; ++ks)
        s[nt] = wmma_bf16(qf[ks], load_b_frag(&k_lds[nt * 16][ks * 32], 128, lane),
                          s[nt]);

    if (kb >= 2 * qb) {  // diagonal region: elementwise causal mask
#pragma unroll
      for (int nt = 0; nt < 4; ++nt)
#pragma unroll
        for (int r = 0; r < 8; ++r) {
          int row = qb * 128 + w * 16 + r + ((lane >> 4) << 3);
          int col = kb * 64 + nt * 16 + (lane & 15);
          s[nt][r] = (col > row) ? -1e10f : s[nt][r];
        }
    }

    // online softmax: rows live in 16-lane halves -> shfl_xor(8,4,2,1)
    float mnew[8], alpha[8];
#pragma unroll
    for (int r = 0; r < 8; ++r) {
      float mx = fmaxf(fmaxf(s[0][r], s[1][r]), fmaxf(s[2][r], s[3][r]));
      mx = fmaxf(mx, __shfl_xor(mx, 8, 32));
      mx = fmaxf(mx, __shfl_xor(mx, 4, 32));
      mx = fmaxf(mx, __shfl_xor(mx, 2, 32));
      mx = fmaxf(mx, __shfl_xor(mx, 1, 32));
      mnew[r] = fmaxf(mrun[r], mx);
      alpha[r] = __expf(mrun[r] - mnew[r]);
      mrun[r] = mnew[r];
    }
#pragma unroll
    for (int nt = 0; nt < 4; ++nt)
#pragma unroll
      for (int r = 0; r < 8; ++r) s[nt][r] = __expf(s[nt][r] - mnew[r]);
#pragma unroll
    for (int r = 0; r < 8; ++r) {
      float sum = s[0][r] + s[1][r] + s[2][r] + s[3][r];
      sum += __shfl_xor(sum, 8, 32);
      sum += __shfl_xor(sum, 4, 32);
      sum += __shfl_xor(sum, 2, 32);
      sum += __shfl_xor(sum, 1, 32);
      lrun[r] = lrun[r] * alpha[r] + sum;
    }
#pragma unroll
    for (int dn = 0; dn < 8; ++dn)
#pragma unroll
      for (int r = 0; r < 8; ++r) oacc[dn][r] *= alpha[r];

    // P -> bf16 -> LDS (re-form A-matrix fragments)
#pragma unroll
    for (int nt = 0; nt < 4; ++nt)
#pragma unroll
      for (int r = 0; r < 8; ++r)
        p_lds[w][r + ((lane >> 4) << 3)][nt * 16 + (lane & 15)] = f2bf(s[nt][r]);
    __syncthreads();

    // O += P @ V : K = 64 keys = 2 WMMA k-steps
#pragma unroll
    for (int ks = 0; ks < 2; ++ks) {
      bf16x16 pf = load_a_frag(&p_lds[w][0][ks * 32], 64, lane);
#pragma unroll
      for (int dn = 0; dn < 8; ++dn)
        oacc[dn] = wmma_bf16(pf, load_b_frag(&vT_lds[dn * 16][ks * 32], 64, lane),
                             oacc[dn]);
    }
    __syncthreads();
  }

  float inv[8];
#pragma unroll
  for (int r = 0; r < 8; ++r) inv[r] = 1.f / lrun[r];
#pragma unroll
  for (int dn = 0; dn < 8; ++dn)
#pragma unroll
    for (int r = 0; r < 8; ++r) {
      int rl = w * 16 + r + ((lane >> 4) << 3);
      int col = dn * 16 + (lane & 15);
      o[((long)(b * 2048 + qb * 128 + rl)) * 2048 + h * 128 + col] =
          f2bf(oacc[dn][r] * inv[r]);
    }
}

// ---------------------------------------------------------------------------
// Host launcher
// ---------------------------------------------------------------------------
extern "C" void kernel_launch(void* const* d_in, const int* in_sizes, int n_in,
                              void* d_out, int out_size, void* d_ws, size_t ws_size,
                              hipStream_t stream) {
  (void)in_sizes; (void)n_in; (void)out_size; (void)ws_size;
  const float* x   = (const float*)d_in[0];  // [2,2048,2048]
  const float* Wf  = (const float*)d_in[1];  // [2048,18688]
  const float* Wao = (const float*)d_in[2];  // [2048,2048]
  const float* Wff = (const float*)d_in[3];  // [8192,2048]
  float* out = (float*)d_out;                // [2,2048,2048]

  char* ws = (char*)d_ws;
  size_t off = 0;
  auto alloc = [&](size_t bytes) {
    char* p = ws + off;
    off += (bytes + 255) & ~(size_t)255;
    return p;
  };
  u16*   xb     = (u16*)alloc(4096ull * 2048 * 2);     // x bf16
  u16*   Wfb    = (u16*)alloc(2048ull * 18688 * 2);    // W_fused bf16
  u16*   Waob   = (u16*)alloc(2048ull * 2048 * 2);     // W_attn_out bf16
  u16*   Wffb   = (u16*)alloc(8192ull * 2048 * 2);     // W_ff_out bf16
  float* pqkv   = (float*)alloc(4096ull * 2304 * 4);   // q,k,v proj (f32)
  u16*   pff    = (u16*)alloc(4096ull * 16384 * 2);    // ff proj (bf16)
  u16*   q_rot  = (u16*)alloc(2ull * 16 * 2048 * 128 * 2);
  u16*   k_rot  = (u16*)alloc(2ull * 2048 * 128 * 2);
  u16*   v_b    = (u16*)alloc(2ull * 2048 * 128 * 2);
  u16*   o_b    = (u16*)alloc(4096ull * 2048 * 2);
  u16*   ff_act = (u16*)alloc(4096ull * 8192 * 2);

  // 1. fp32 -> bf16 conversions
  cvt_f32_bf16<<<4096, 256, 0, stream>>>(x,   xb,   4096l * 2048);
  cvt_f32_bf16<<<4096, 256, 0, stream>>>(Wf,  Wfb,  2048l * 18688);
  cvt_f32_bf16<<<4096, 256, 0, stream>>>(Wao, Waob, 2048l * 2048);
  cvt_f32_bf16<<<4096, 256, 0, stream>>>(Wff, Wffb, 8192l * 2048);

  // 2. fused projection split into qkv (f32) and ff (bf16) column blocks
  gemm_bf16<0><<<dim3(2304 / 128, 4096 / 128), 256, 0, stream>>>(
      xb, Wfb, pqkv, 2048, 18688, 2304);
  gemm_bf16<1><<<dim3(16384 / 128, 4096 / 128), 256, 0, stream>>>(
      xb, Wfb + 2304, pff, 2048, 18688, 16384);

  // 3. xpos rotary + q-scale; k,v to bf16
  rotary_kernel<<<dim3(2048, 2), 128, 0, stream>>>(pqkv, q_rot, k_rot, v_b);

  // 4. causal multi-query flash attention
  attn_kernel<<<dim3(16, 16, 2), 256, 0, stream>>>(q_rot, k_rot, v_b, o_b);

  // 5. attn_out = o @ W_attn_out  (writes d_out)
  gemm_bf16<0><<<dim3(2048 / 128, 4096 / 128), 256, 0, stream>>>(
      o_b, Waob, out, 2048, 2048, 2048);

  // 6. SiLU-gated ff activation
  silu_mul<<<8192, 256, 0, stream>>>(pff, ff_act);

  // 7. d_out += ff_act @ W_ff_out
  gemm_bf16<2><<<dim3(2048 / 128, 4096 / 128), 256, 0, stream>>>(
      ff_act, Wffb, out, 8192, 2048, 2048);
}
